// SlotPathF_44032004718740
// MI455X (gfx1250) — matrix operand order repaired and checked by reading, not compiled, not run on hardware
//
#include <hip/hip_runtime.h>
#include <math.h>

// ---------------- types ----------------
typedef unsigned short u16;
typedef unsigned int   u32;
typedef __attribute__((ext_vector_type(16))) __bf16 v16bf;
typedef __attribute__((ext_vector_type(8)))  float  v8f;

union FragB { v16bf v; u32 u[8]; };

static __device__ __forceinline__ u16 f2bf(float f) {
    u32 u = __float_as_uint(f);
    u += 0x7FFFu + ((u >> 16) & 1u);          // round-to-nearest-even
    return (u16)(u >> 16);
}
static __device__ __forceinline__ float gelu_f(float v) {
    return 0.5f * v * (1.0f + erff(v * 0.70710678118654752f));
}
static __device__ __forceinline__ float sigm_f(float v) {
    return 1.0f / (1.0f + expf(-v));
}

// problem dims
#define BB 4
#define TT 4096
#define DD 1024
#define HH 16
#define HDD 64
#define NSS 64
#define HMM 4096
#define MM (BB*TT)     // 16384

// ---------------- fp32 -> bf16 weight conversion ----------------
__global__ void f32_to_bf16_k(const float* __restrict__ in, u16* __restrict__ out, int n) {
    int i = blockIdx.x * 256 + threadIdx.x;
    if (i < n) out[i] = f2bf(in[i]);
}

// ---------------- bias1 = rb1 + slot_mean_full @ rw1[D:2D] ----------------
__global__ __launch_bounds__(256) void bias1_k(const float* __restrict__ slot_init,
                                               const float* __restrict__ rw1,
                                               const float* __restrict__ rb1,
                                               float* __restrict__ bias1) {
    __shared__ float sm[64];
    int tid = threadIdx.x;
    if (tid < 64) {
        float a = 0.f;
        for (int s = 0; s < 64; ++s) a += slot_init[s * 64 + tid];
        sm[tid] = a * (1.0f / 64.0f);
    }
    __syncthreads();
    int n = blockIdx.x * 256 + tid;          // grid = 4 blocks -> n in [0,1024)
    float acc = rb1[n];
    for (int d = 0; d < DD; ++d)
        acc += sm[d & 63] * rw1[(size_t)(DD + d) * DD + n];
    bias1[n] = acc;
}

// ---------------- WMMA GEMM: out = act(A[M,K] @ Wbf16[K,N] + bias[N]) ----------------
// Block tile 128(M) x 64(N), BK = 32, 256 threads = 8 waves in a 4x2 grid.
// Each wave owns a 32x32 macro-tile: 2 A frags x 2 B frags -> 4 v_wmma per K-step
// (8 ds_load_b32 per WMMA instead of 12 with a 16x32 wave tile).
template<bool A_F32, bool GELU, bool OUT_BF16>
__global__ __launch_bounds__(256) void wmma_gemm_k(const void* __restrict__ A,
                                                   const u16* __restrict__ W,
                                                   const float* __restrict__ bias,
                                                   void* __restrict__ out,
                                                   int N, int K) {
    __shared__ u16 lA[128 * 34];  // [m][k], row padded to 34 elems (68B): 4B-aligned u32 reads, bank-spread
    __shared__ u16 lB[64 * 34];   // [n][k] transposed so fragment k-pairs are contiguous

    const int tid  = threadIdx.x;
    const int lane = tid & 31;
    const int w    = tid >> 5;
    const int wr   = w >> 1;          // wave row 0..3  -> 32 rows each
    const int wc   = w & 1;           // wave col 0..1  -> 32 cols each
    const int n0   = blockIdx.x * 64;
    const int m0   = blockIdx.y * 128;

    v8f acc00 = {0.f,0.f,0.f,0.f,0.f,0.f,0.f,0.f};
    v8f acc01 = acc00, acc10 = acc00, acc11 = acc00;

    const int half = lane >> 4;
    const int l15  = lane & 15;
    const int mr0  = wr * 32 + l15;        // A frag rows
    const int mr1  = mr0 + 16;
    const int ncA  = wc * 32 + l15;        // B frag cols
    const int ncB  = ncA + 16;

    // staging maps
    const int srow = tid >> 1, scg = tid & 1;       // A: 16 elems (row, cols scg*16..)
    const int bkk  = tid >> 3, bng = tid & 7;       // B: 8 elems  (k-row, cols bng*8..)

    for (int k0 = 0; k0 < K; k0 += 32) {
        // ---- prefetch next K tile into cache while this one computes ----
        if (k0 + 32 < K) {
            if constexpr (A_F32)
                __builtin_prefetch((const float*)A + (size_t)(m0 + srow) * K + k0 + 32 + scg * 16, 0, 1);
            else
                __builtin_prefetch((const u16*)A + (size_t)(m0 + srow) * K + k0 + 32 + scg * 16, 0, 1);
            __builtin_prefetch(W + (size_t)(k0 + 32 + bkk) * N + n0 + bng * 8, 0, 1);
        }
        // ---- stage A tile 128x32 (16 elems/thread) ----
        {
            u32 pk[8];
            if constexpr (A_F32) {
                const float* ap = (const float*)A + (size_t)(m0 + srow) * K + k0 + scg * 16;
                #pragma unroll
                for (int q = 0; q < 4; ++q) {
                    float4 f = ((const float4*)ap)[q];
                    pk[2 * q]     = (u32)f2bf(f.x) | ((u32)f2bf(f.y) << 16);
                    pk[2 * q + 1] = (u32)f2bf(f.z) | ((u32)f2bf(f.w) << 16);
                }
            } else {
                const u16* ap = (const u16*)A + (size_t)(m0 + srow) * K + k0 + scg * 16;
                uint4 r0 = ((const uint4*)ap)[0];
                uint4 r1 = ((const uint4*)ap)[1];
                pk[0] = r0.x; pk[1] = r0.y; pk[2] = r0.z; pk[3] = r0.w;
                pk[4] = r1.x; pk[5] = r1.y; pk[6] = r1.z; pk[7] = r1.w;
            }
            u32* dst = (u32*)&lA[srow * 34 + scg * 16];
            #pragma unroll
            for (int q = 0; q < 8; ++q) dst[q] = pk[q];
        }
        // ---- stage B tile 32x64 transposed into lB[n][k] ----
        {
            const u16* bp = W + (size_t)(k0 + bkk) * N + n0 + bng * 8;
            uint4 raw = *(const uint4*)bp;
            u32 rr[4] = {raw.x, raw.y, raw.z, raw.w};
            #pragma unroll
            for (int e = 0; e < 8; ++e)
                lB[(bng * 8 + e) * 34 + bkk] = (u16)(rr[e >> 1] >> ((e & 1) * 16));
        }
        __syncthreads();

        // ---- build fragments (ISA 7.12.2 16-bit layouts) ----
        FragB fa0, fa1, fb0, fb1;
        #pragma unroll
        for (int j = 0; j < 8; ++j) {
            int k2 = (j < 4) ? (half * 8 + 2 * j) : (16 + half * 8 + 2 * (j - 4));
            fa0.u[j] = *(const u32*)&lA[mr0 * 34 + k2];
            fa1.u[j] = *(const u32*)&lA[mr1 * 34 + k2];
        }
        #pragma unroll
        for (int j = 0; j < 8; ++j) {
            int k2 = (j < 4) ? (half * 8 + 2 * j) : (16 + half * 8 + 2 * (j - 4));
            fb0.u[j] = *(const u32*)&lB[ncA * 34 + k2];
            fb1.u[j] = *(const u32*)&lB[ncB * 34 + k2];
        }
        acc00 = __builtin_amdgcn_wmma_f32_16x16x32_bf16(false, fa0.v, false, fb0.v, (short)0, acc00, false, false);
        acc01 = __builtin_amdgcn_wmma_f32_16x16x32_bf16(false, fa0.v, false, fb1.v, (short)0, acc01, false, false);
        acc10 = __builtin_amdgcn_wmma_f32_16x16x32_bf16(false, fa1.v, false, fb0.v, (short)0, acc10, false, false);
        acc11 = __builtin_amdgcn_wmma_f32_16x16x32_bf16(false, fa1.v, false, fb1.v, (short)0, acc11, false, false);
        __syncthreads();
    }

    // ---- epilogue: C layout lane,(r): row = r + 8*(lane>=16), col = lane&15 ----
    const int gcA = n0 + ncA;
    const int gcB = n0 + ncB;
    const float bvA = bias[gcA];
    const float bvB = bias[gcB];
    const int gmBase0 = m0 + wr * 32;
    #pragma unroll
    for (int r = 0; r < 8; ++r) {
        int gm0 = gmBase0 + r + (half << 3);
        int gm1 = gm0 + 16;
        float v00 = acc00[r] + bvA, v01 = acc01[r] + bvB;
        float v10 = acc10[r] + bvA, v11 = acc11[r] + bvB;
        if constexpr (GELU) {
            v00 = gelu_f(v00); v01 = gelu_f(v01);
            v10 = gelu_f(v10); v11 = gelu_f(v11);
        }
        if constexpr (OUT_BF16) {
            u16* o = (u16*)out;
            o[(size_t)gm0 * N + gcA] = f2bf(v00);
            o[(size_t)gm0 * N + gcB] = f2bf(v01);
            o[(size_t)gm1 * N + gcA] = f2bf(v10);
            o[(size_t)gm1 * N + gcB] = f2bf(v11);
        } else {
            float* o = (float*)out;
            o[(size_t)gm0 * N + gcA] = v00;
            o[(size_t)gm0 * N + gcB] = v01;
            o[(size_t)gm1 * N + gcA] = v10;
            o[(size_t)gm1 * N + gcB] = v11;
        }
    }
}

// ---------------- top-32-of-64 + softmax -> dense alpha ----------------
__global__ __launch_bounds__(256) void topk_k(const float* __restrict__ logits,
                                              const float* __restrict__ tau,
                                              float* __restrict__ alpha) {
    __shared__ float lv[8][64];
    int w = threadIdx.x >> 5, lane = threadIdx.x & 31;
    int token = blockIdx.x * 8 + w;
    float invt = 1.0f / (fabsf(tau[0]) + 0.1f);
    float v0 = logits[(size_t)token * 64 + lane] * invt;
    float v1 = logits[(size_t)token * 64 + 32 + lane] * invt;
    lv[w][lane] = v0; lv[w][lane + 32] = v1;
    __syncthreads();
    int r0 = 0, r1 = 0;
    #pragma unroll 4
    for (int i = 0; i < 64; ++i) {
        float u = lv[w][i];
        r0 += (u > v0) || (u == v0 && i < lane);
        r1 += (u > v1) || (u == v1 && i < lane + 32);
    }
    bool s0 = r0 < 32, s1 = r1 < 32;
    float m = fmaxf(v0, v1);
    for (int off = 16; off; off >>= 1) m = fmaxf(m, __shfl_xor(m, off, 32));
    float e0 = s0 ? expf(v0 - m) : 0.f;
    float e1 = s1 ? expf(v1 - m) : 0.f;
    float sum = e0 + e1;
    for (int off = 16; off; off >>= 1) sum += __shfl_xor(sum, off, 32);
    float inv = 1.0f / sum;
    alpha[(size_t)token * 64 + lane]      = e0 * inv;
    alpha[(size_t)token * 64 + 32 + lane] = e1 * inv;
}

// ---------------- pooling: slot_in[b,h,s,:] = (sum_t a*x) / (sum_t a + 1e-8) ----------------
__global__ __launch_bounds__(256) void pool_k(const float* __restrict__ x,
                                              const float* __restrict__ alpha,
                                              float* __restrict__ slot_in) {
    __shared__ float la[64 * 4];
    __shared__ float lx[64 * 64];
    int bh = blockIdx.x, b = bh >> 4, h = bh & 15;
    int tid = threadIdx.x, s = tid >> 6, d = tid & 63;
    float acc = 0.f, accw = 0.f;
    for (int t0 = 0; t0 < TT; t0 += 64) {
        { int i = tid >> 2, sp = tid & 3;
          la[i * 4 + sp] = alpha[((size_t)(b * TT + t0 + i)) * 64 + h * 4 + sp]; }
        #pragma unroll
        for (int r = 0; r < 16; ++r) {
            int i = r * 4 + s;
            lx[i * 64 + d] = x[((size_t)(b * TT + t0 + i)) * DD + h * 64 + d];
        }
        __syncthreads();
        #pragma unroll 8
        for (int i = 0; i < 64; ++i) {
            float a = la[i * 4 + s];
            acc  += a * lx[i * 64 + d];
            accw += a;
        }
        __syncthreads();
    }
    slot_in[((size_t)bh * 4 + s) * 64 + d] = acc / (accw + 1e-8f);
}

// ---------------- GRU cell + hidden/out proj (one block per slot) ----------------
__global__ __launch_bounds__(64) void gru_k(const float* __restrict__ slot_in,
                                            const float* __restrict__ slot_init,
                                            const float* __restrict__ gwi, const float* __restrict__ gwh,
                                            const float* __restrict__ gbi, const float* __restrict__ gbh,
                                            const float* __restrict__ hpw, const float* __restrict__ hpb,
                                            const float* __restrict__ ohw, const float* __restrict__ ohb,
                                            float* __restrict__ S_new) {
    __shared__ float sin_[64], hprev[64], hnew[64], hidden[256];
    int j = threadIdx.x, bhs = blockIdx.x, hs = bhs & 63;
    sin_[j]  = slot_in[(size_t)bhs * 64 + j];
    hprev[j] = slot_init[hs * 64 + j];
    __syncthreads();
    float gi[3], gh[3];
    #pragma unroll
    for (int g = 0; g < 3; ++g) {
        int r = g * 64 + j;
        float ai = gbi[r], ah = gbh[r];
        const float* wi = gwi + (size_t)r * 64;
        const float* wh = gwh + (size_t)r * 64;
        for (int d = 0; d < 64; ++d) { ai += wi[d] * sin_[d]; ah += wh[d] * hprev[d]; }
        gi[g] = ai; gh[g] = ah;
    }
    float rg = sigm_f(gi[0] + gh[0]);
    float zg = sigm_f(gi[1] + gh[1]);
    float ng = tanhf(gi[2] + rg * gh[2]);
    hnew[j] = (1.0f - zg) * ng + zg * hprev[j];
    __syncthreads();
    #pragma unroll
    for (int g = 0; g < 4; ++g) {
        int c = g * 64 + j;
        float a = hpb[c];
        for (int d = 0; d < 64; ++d) a += hnew[d] * hpw[(size_t)d * 256 + c];
        hidden[c] = gelu_f(a);
    }
    __syncthreads();
    float a = ohb[j];
    for (int c = 0; c < 256; ++c) a += hidden[c] * ohw[(size_t)c * 64 + j];
    S_new[(size_t)bhs * 64 + j] = a;
}

// ---------------- broadcast back to tokens: out1 = einsum(alpha, S_new) (bf16) ----------------
__global__ __launch_bounds__(256) void bcast_k(const float* __restrict__ alpha,
                                               const float* __restrict__ S_new,
                                               u16* __restrict__ out1) {
    __shared__ float lS[4096];
    __shared__ float la[1024];
    int blk = blockIdx.x, b = blk >> 8, t0 = (blk & 255) * 16;
    int tid = threadIdx.x;
    #pragma unroll
    for (int r = 0; r < 16; ++r) lS[r * 256 + tid] = S_new[(size_t)b * 4096 + r * 256 + tid];
    #pragma unroll
    for (int r = 0; r < 4; ++r) {
        int li = r * 256 + tid;
        la[li] = alpha[((size_t)(b * TT + t0 + (li >> 6))) * 64 + (li & 63)];
    }
    __syncthreads();
    for (int tl = 0; tl < 16; ++tl) {
        #pragma unroll
        for (int rep = 0; rep < 4; ++rep) {
            int col = rep * 256 + tid, h = col >> 6, dd = col & 63;
            const float* ap = &la[tl * 64 + h * 4];
            float v = ap[0] * lS[(h * 4 + 0) * 64 + dd] + ap[1] * lS[(h * 4 + 1) * 64 + dd]
                    + ap[2] * lS[(h * 4 + 2) * 64 + dd] + ap[3] * lS[(h * 4 + 3) * 64 + dd];
            out1[((size_t)(b * TT + t0 + tl)) * DD + col] = f2bf(v);
        }
    }
}

// ---------------- launch ----------------
extern "C" void kernel_launch(void* const* d_in, const int* in_sizes, int n_in,
                              void* d_out, int out_size, void* d_ws, size_t ws_size,
                              hipStream_t stream) {
    (void)in_sizes; (void)n_in; (void)out_size; (void)ws_size;
    const float* x         = (const float*)d_in[0];
    const float* slot_init = (const float*)d_in[1];
    const float* rw1 = (const float*)d_in[2];
    const float* rb1 = (const float*)d_in[3];
    const float* rw2 = (const float*)d_in[4];
    const float* rb2 = (const float*)d_in[5];
    const float* gwi = (const float*)d_in[6];
    const float* gwh = (const float*)d_in[7];
    const float* gbi = (const float*)d_in[8];
    const float* gbh = (const float*)d_in[9];
    const float* hpw = (const float*)d_in[10];
    const float* hpb = (const float*)d_in[11];
    const float* ohw = (const float*)d_in[12];
    const float* ohb = (const float*)d_in[13];
    const float* vpw = (const float*)d_in[14];
    const float* vpb = (const float*)d_in[15];
    const float* vow = (const float*)d_in[16];
    const float* vob = (const float*)d_in[17];
    const float* opw = (const float*)d_in[18];
    const float* opb = (const float*)d_in[19];
    const float* tau = (const float*)d_in[20];

    char* p = (char*)d_ws;
    auto alloc = [&](size_t bytes) -> void* {
        void* r = (void*)p;
        p += (bytes + 255) & ~(size_t)255;
        return r;
    };
    u16*   wrw1   = (u16*)  alloc((size_t)DD * DD * 2);         // rw1 top half bf16
    u16*   wrw2   = (u16*)  alloc((size_t)DD * NSS * 2);
    u16*   wvpw   = (u16*)  alloc((size_t)DD * HMM * 2);
    u16*   wvow   = (u16*)  alloc((size_t)HMM * DD * 2);
    u16*   wopw   = (u16*)  alloc((size_t)DD * DD * 2);
    float* bias1  = (float*)alloc((size_t)DD * 4);
    u16*   h1     = (u16*)  alloc((size_t)MM * DD * 2);
    float* logits = (float*)alloc((size_t)MM * NSS * 4);
    float* alpha  = (float*)alloc((size_t)MM * NSS * 4);
    float* slotin = (float*)alloc((size_t)BB * HH * 4 * HDD * 4);
    float* Snew   = (float*)alloc((size_t)BB * HH * 4 * HDD * 4);
    u16*   out1   = (u16*)  alloc((size_t)MM * DD * 2);
    u16*   act3   = (u16*)  alloc((size_t)MM * HMM * 2);
    u16*   out2   = (u16*)  alloc((size_t)MM * DD * 2);

    // weight conversions (rw1: only top D rows -> contiguous first D*D elems)
    f32_to_bf16_k<<<(DD * DD + 255) / 256, 256, 0, stream>>>(rw1, wrw1, DD * DD);
    f32_to_bf16_k<<<(DD * NSS + 255) / 256, 256, 0, stream>>>(rw2, wrw2, DD * NSS);
    f32_to_bf16_k<<<(DD * HMM + 255) / 256, 256, 0, stream>>>(vpw, wvpw, DD * HMM);
    f32_to_bf16_k<<<(HMM * DD + 255) / 256, 256, 0, stream>>>(vow, wvow, HMM * DD);
    f32_to_bf16_k<<<(DD * DD + 255) / 256, 256, 0, stream>>>(opw, wopw, DD * DD);

    // constant-folded routing bias
    bias1_k<<<4, 256, 0, stream>>>(slot_init, rw1, rb1, bias1);

    // G1: h1 = gelu(x @ rw1_top + bias1)   [A fp32 -> bf16 WMMA]
    wmma_gemm_k<true, true, true><<<dim3(DD / 64, MM / 128), 256, 0, stream>>>(
        (const void*)x, wrw1, bias1, (void*)h1, DD, DD);
    // G2: logits = h1 @ rw2 + rb2 (fp32 out; tau scaling applied in topk)
    wmma_gemm_k<false, false, false><<<dim3(NSS / 64, MM / 128), 256, 0, stream>>>(
        (const void*)h1, wrw2, rb2, (void*)logits, NSS, DD);

    topk_k<<<MM / 8, 256, 0, stream>>>(logits, tau, alpha);
    pool_k<<<BB * HH, 256, 0, stream>>>(x, alpha, slotin);
    gru_k<<<BB * HH * 4, 64, 0, stream>>>(slotin, slot_init, gwi, gwh, gbi, gbh,
                                          hpw, hpb, ohw, ohb, Snew);
    bcast_k<<<BB * (TT / 16), 256, 0, stream>>>(alpha, Snew, out1);

    // G3: act3 = gelu(out1 @ vpw + vpb)
    wmma_gemm_k<false, true, true><<<dim3(HMM / 64, MM / 128), 256, 0, stream>>>(
        (const void*)out1, wvpw, vpb, (void*)act3, HMM, DD);
    // G4: out2 = act3 @ vow + vob
    wmma_gemm_k<false, false, true><<<dim3(DD / 64, MM / 128), 256, 0, stream>>>(
        (const void*)act3, wvow, vob, (void*)out2, DD, HMM);
    // G5: out = out2 @ opw + opb (fp32 to d_out)
    wmma_gemm_k<false, false, false><<<dim3(DD / 64, MM / 128), 256, 0, stream>>>(
        (const void*)out2, wopw, opb, d_out, DD, DD);
}
